// NaryTreeGRUCell_16441134809400
// MI455X (gfx1250) — matrix operand order repaired and checked by reading, not compiled, not run on hardware
//
#include <hip/hip_runtime.h>
#include <hip/hip_bf16.h>

// ---------------- types for WMMA fragments ----------------
typedef __attribute__((ext_vector_type(16))) __bf16 v16bf;
typedef __attribute__((ext_vector_type(8)))  __bf16 v8bf;
typedef __attribute__((ext_vector_type(2)))  __bf16 v2bf;
typedef __attribute__((ext_vector_type(8)))  float  v8f;

#define BM   64          // node rows per workgroup
#define LDA  1296        // LDS row stride in bf16 elements (padded: 2592B % 256 != 0)
#define NT   512         // threads per workgroup (16 waves)
#define XS   256
#define HC   1024        // n_ary * h
#define KTOT 1280

// offsets (in bf16 elements) inside d_ws for converted weights
#define W_ELEMS   (768 * 256)       // 196608
#define U_ELEMS   (768 * 1024)      // 786432
#define U2_ELEMS  (256 * 1024)      // 262144
#define U_OFF     (W_ELEMS)                 // 196608
#define U2_OFF    (W_ELEMS + U_ELEMS)       // 983040
#define TOT_ELEMS (W_ELEMS + U_ELEMS + U2_ELEMS)  // 1245184

// Packed fp32x2 -> bf16x2 conversion (1 VALU op instead of ~8).
__device__ __forceinline__ unsigned int pack2bf(float a, float b) {
#if __has_builtin(__builtin_amdgcn_cvt_pk_bf16_f32)
    v2bf r = __builtin_amdgcn_cvt_pk_bf16_f32(a, b);
    return __builtin_bit_cast(unsigned int, r);
#else
    unsigned int r;
    asm("v_cvt_pk_bf16_f32 %0, %1, %2" : "=v"(r) : "v"(a), "v"(b));
    return r;
#endif
}

__device__ __forceinline__ unsigned short f2bf(float f) {
    unsigned int x = __float_as_uint(f);
    unsigned int r = x + 0x7FFFu + ((x >> 16) & 1u);   // round-to-nearest-even
    return (unsigned short)(r >> 16);
}

// ---------------- weight fp32 -> bf16 pre-pass (pairwise, packed cvt) ----------------
__global__ void cvt_weights_kernel(const float* __restrict__ W,
                                   const float* __restrict__ U,
                                   const float* __restrict__ U2,
                                   unsigned int* __restrict__ ws32) {
    unsigned int p = blockIdx.x * blockDim.x + threadIdx.x;   // pair index
    unsigned int i = p * 2;
    if (i >= TOT_ELEMS) return;
    float a, b;
    if (i < W_ELEMS)      { a = W[i];            b = W[i + 1]; }
    else if (i < U2_OFF)  { a = U[i - U_OFF];    b = U[i - U_OFF + 1]; }
    else                  { a = U2[i - U2_OFF];  b = U2[i - U2_OFF + 1]; }
    ws32[p] = pack2bf(a, b);
}

// ---------------- fragment loaders (contiguous k order, same for A and B) ----------------
// A 16x32 bf16: lane L (r=L&15, g=L>>4) holds row r, K = g*16 + {0..15}
__device__ __forceinline__ v16bf load_a_frag(const unsigned short* As, int row,
                                             int k, int lane) {
    const int koff = (lane >> 4) << 4;                       // 0 or 16
    const unsigned short* p = As + row * LDA + k + koff;
    v8bf lo = *(const v8bf*)p;                               // K = k+koff .. +7
    v8bf hi = *(const v8bf*)(p + 8);                         // K = k+8+koff .. +7
    return __builtin_shufflevector(lo, hi, 0,1,2,3,4,5,6,7,8,9,10,11,12,13,14,15);
}

// B 32x16 bf16: lane L holds column n=L&15, K = (L>>4)*16 + {0..15}.
// Weights are [out][K] row-major -> 16 consecutive K = one 32B contiguous load.
__device__ __forceinline__ v16bf load_b_frag(const unsigned short* Wg, int n,
                                             int ldb, int k, int lane) {
    const int col = lane & 15;
    const int kg  = (lane >> 4) << 4;                        // 0 or 16
    return *(const v16bf*)(Wg + (size_t)(n + col) * ldb + k + kg);
}

__device__ __forceinline__ v8f wmma_bf16(v16bf a, v16bf b, v8f c) {
    return __builtin_amdgcn_wmma_f32_16x16x32_bf16(
        /*neg_a=*/false, a, /*neg_b=*/false, b,
        /*c_mod=*/(short)0, c, /*reuse_a=*/false, /*reuse_b=*/false);
}

// ---------------- fused TreeGRU tile kernel ----------------
__global__ void __launch_bounds__(NT, 1)
treegru_tile_kernel(const float* __restrict__ x,
                    const float* __restrict__ h_prev,
                    const int*   __restrict__ child_idx,
                    const int*   __restrict__ child_mask,
                    const float* __restrict__ b_ruo,
                    const unsigned short* __restrict__ Wbf,   // [768][256]
                    const unsigned short* __restrict__ Ubf,   // [768][1024]
                    const unsigned short* __restrict__ U2bf,  // [256][1024]
                    float* __restrict__ out,
                    int N) {
    extern __shared__ unsigned short As[];   // [BM][LDA] bf16
    const int tid = threadIdx.x;
    const int m0  = blockIdx.x * BM;

    // ---- Phase 0a: x tile -> As[:, 0:256] (fp32 -> bf16, packed cvt) ----
    for (int i = tid; i < (BM * XS / 4); i += NT) {
        int r  = i >> 6;                 // 64 float4 per row
        int c4 = i & 63;
        int rr = m0 + r; if (rr >= N) rr = N - 1;
        float4 v = ((const float4*)(x + (size_t)rr * XS))[c4];
        unsigned int* d = (unsigned int*)(As + r * LDA + c4 * 4);
        d[0] = pack2bf(v.x, v.y);
        d[1] = pack2bf(v.z, v.w);
    }

    // ---- Phase 0b: masked child gather -> As[:, 256:1280] ----
    // 256 (row,child) pairs; 2 threads per pair, 128 floats each
    {
        int pair = tid >> 1;             // 0..255
        int half = tid & 1;
        int r = pair >> 2;               // tile row
        int c = pair & 3;                // child
        int rr = m0 + r; if (rr >= N) rr = N - 1;
        int idx = child_idx [(size_t)rr * 4 + c];
        int msk = child_mask[(size_t)rr * 4 + c];
        if (idx < 0) idx = 0; if (idx >= N) idx = N - 1;
        unsigned short* dst = As + r * LDA + XS + c * 256 + half * 128;
        if (msk) {
            const float4* src = (const float4*)(h_prev + (size_t)idx * 256) + half * 32;
            for (int j = 0; j < 32; ++j) {
                float4 v = src[j];
                ((unsigned int*)dst)[2 * j]     = pack2bf(v.x, v.y);
                ((unsigned int*)dst)[2 * j + 1] = pack2bf(v.z, v.w);
            }
        } else {
            for (int j = 0; j < 64; ++j) ((unsigned int*)dst)[j] = 0u;
        }
    }
    __syncthreads();

    // ---- Phase 1: WMMA over K, three gates (u, o, h_inner) ----
    const int lane = tid & 31;
    const int wave = tid >> 5;          // 0..15
    const int rowb = (wave & 1) * 32;   // M base within tile
    const int colb = (wave >> 1) * 32;  // N base within 256 out columns

    v8f z = {0.f, 0.f, 0.f, 0.f, 0.f, 0.f, 0.f, 0.f};
    v8f au[2][2], ao[2][2], ah[2][2];
    for (int a = 0; a < 2; ++a)
        for (int b = 0; b < 2; ++b) { au[a][b] = z; ao[a][b] = z; ah[a][b] = z; }

    const int r0 = rowb + (lane & 15);
    const int r1 = rowb + 16 + (lane & 15);

    // K segment 1: x part (k = 0..255) — feeds u and o gates only
    for (int k = 0; k < XS; k += 32) {
        v16bf a0 = load_a_frag(As, r0, k, lane);
        v16bf a1 = load_a_frag(As, r1, k, lane);
        #pragma unroll
        for (int ns = 0; ns < 2; ++ns) {
            int n = colb + ns * 16;
            v16bf bu = load_b_frag(Wbf + 256 * 256, n, 256, k, lane); // W_u rows 256..511
            au[0][ns] = wmma_bf16(a0, bu, au[0][ns]);
            au[1][ns] = wmma_bf16(a1, bu, au[1][ns]);
            v16bf bo = load_b_frag(Wbf + 512 * 256, n, 256, k, lane); // W_o rows 512..767
            ao[0][ns] = wmma_bf16(a0, bo, ao[0][ns]);
            ao[1][ns] = wmma_bf16(a1, bo, ao[1][ns]);
        }
    }

    // K segment 2: h_cat part (k = 0..1023) — feeds u, o, h_inner
    for (int k = 0; k < HC; k += 32) {
        v16bf a0 = load_a_frag(As, r0, XS + k, lane);
        v16bf a1 = load_a_frag(As, r1, XS + k, lane);
        #pragma unroll
        for (int ns = 0; ns < 2; ++ns) {
            int n = colb + ns * 16;
            v16bf bu = load_b_frag(Ubf + 256 * 1024, n, 1024, k, lane); // U_u
            au[0][ns] = wmma_bf16(a0, bu, au[0][ns]);
            au[1][ns] = wmma_bf16(a1, bu, au[1][ns]);
            v16bf bo = load_b_frag(Ubf + 512 * 1024, n, 1024, k, lane); // U_o
            ao[0][ns] = wmma_bf16(a0, bo, ao[0][ns]);
            ao[1][ns] = wmma_bf16(a1, bo, ao[1][ns]);
            v16bf bh = load_b_frag(U2bf, n, 1024, k, lane);             // U_u2
            ah[0][ns] = wmma_bf16(a0, bh, ah[0][ns]);
            ah[1][ns] = wmma_bf16(a1, bh, ah[1][ns]);
        }
    }

    // ---- Phase 2: epilogue. C/D layout: N = lane&15, M = vgpr + 8*(lane>>4) ----
    #pragma unroll
    for (int ns = 0; ns < 2; ++ns) {
        int n  = colb + ns * 16 + (lane & 15);
        float bu = b_ruo[256 + n];
        float bo = b_ruo[512 + n];
        #pragma unroll
        for (int ms = 0; ms < 2; ++ms) {
            int mbase = m0 + rowb + ms * 16 + ((lane >> 4) << 3);
            v8f u = au[ms][ns], o = ao[ms][ns], hi = ah[ms][ns];
            #pragma unroll
            for (int j = 0; j < 8; ++j) {
                int row = mbase + j;
                if (row < N) {
                    float uu = 1.0f / (1.0f + __expf(-(u[j] + bu)));
                    float oo = tanhf(o[j] + bo);
                    out[(size_t)row * 256 + n] = oo * uu + (1.0f - uu) * hi[j];
                }
            }
        }
    }
}

// ---------------- launcher ----------------
extern "C" void kernel_launch(void* const* d_in, const int* in_sizes, int n_in,
                              void* d_out, int out_size, void* d_ws, size_t ws_size,
                              hipStream_t stream) {
    const float* x          = (const float*)d_in[0];
    const float* h_prev     = (const float*)d_in[1];
    const float* W_ruo      = (const float*)d_in[2];
    const float* U_ruo      = (const float*)d_in[3];
    const float* b_ruo      = (const float*)d_in[4];
    const float* U_u2       = (const float*)d_in[5];
    const int*   child_idx  = (const int*)d_in[6];
    const int*   child_mask = (const int*)d_in[7];
    float*       out        = (float*)d_out;
    unsigned short* wbf     = (unsigned short*)d_ws;   // ~2.5 MB used

    // 1) weights fp32 -> bf16 into workspace (L2 resident thereafter)
    int npairs = TOT_ELEMS / 2;
    cvt_weights_kernel<<<(npairs + 255) / 256, 256, 0, stream>>>(
        W_ruo, U_ruo, U_u2, (unsigned int*)wbf);

    // 2) fused gather + 3-GEMM + activation
    int N = in_sizes[0] / XS;                     // 200000
    int nblocks = (N + BM - 1) / BM;              // 3125
    size_t lds_bytes = (size_t)BM * LDA * sizeof(unsigned short);  // ~166 KB
    treegru_tile_kernel<<<nblocks, NT, lds_bytes, stream>>>(
        x, h_prev, child_idx, child_mask, b_ruo,
        wbf, wbf + U_OFF, wbf + U2_OFF, out, N);
}
